// ExpertRunner_73933567033636
// MI455X (gfx1250) — matrix operand order repaired and checked by reading, not compiled, not run on hardware
//
#include <hip/hip_runtime.h>

// ---------------------------------------------------------------------------
// Fused MoE (top-2 of 8 experts, SwiGLU) for MI455X (gfx1250, wave32, WMMA).
// T=2048 tokens, D=1024 hidden, F=4096 ffn, E=8 experts, top-k=2.
// Memory-bound: one pass over 402 MB of f32 weights (~17us at 23.3 TB/s).
// Compute: f32 -> f16 conversion in registers during LDS staging (batched
// loads for latency overlap), then v_wmma_f32_16x16x32_f16 with f32
// accumulation. Activations staged to LDS via GLOBAL_LOAD_ASYNC_TO_LDS_B128
// (ASYNCcnt), overlapped with the w2 conversion pipeline.
// ---------------------------------------------------------------------------

#define NEXP 8
#define HID  1024
#define FFN  4096
#define TOK  2048

#define MT   32      // tokens (rows) per block   = 2 wmma M-tiles
#define NT   128     // output cols per block     = 8 wmma N-tiles (2 per wave)
#define KS   128     // K-section staged in LDS per iteration
#define XSTR 136     // LDS row stride (f16) for 128-wide tiles (16B aligned)

#if defined(__has_builtin)
# if __has_builtin(__builtin_amdgcn_global_load_async_to_lds_b128) && \
     __has_builtin(__builtin_amdgcn_s_wait_asynccnt)
#  define HAVE_ASYNC_LDS 1
# endif
#endif
#ifndef HAVE_ASYNC_LDS
# define HAVE_ASYNC_LDS 0
#endif

typedef __attribute__((ext_vector_type(16))) _Float16 v16h;
typedef __attribute__((ext_vector_type(8)))  _Float16 v8h;
typedef __attribute__((ext_vector_type(4)))  _Float16 v4h;
typedef __attribute__((ext_vector_type(8)))  float    v8f;
typedef __attribute__((ext_vector_type(4)))  float    v4f;
typedef __attribute__((ext_vector_type(4)))  int      v4i;

struct Entry { int ts; float w; };   // ts = 2*token + slot, w = combine weight

__device__ __forceinline__ v16h mkv16(v8h lo, v8h hi) {
    v16h r;
#pragma unroll
    for (int i = 0; i < 8; ++i) { r[i] = lo[i]; r[i + 8] = hi[i]; }
    return r;
}

__device__ __forceinline__ v8f wmma_f16(v16h a, v16h b, v8f c) {
    // (neg_a, A, neg_b, B, c_mod, C, reuse_a, reuse_b)
    return __builtin_amdgcn_wmma_f32_16x16x32_f16(false, a, false, b,
                                                  (short)0, c, false, false);
}

// Convert float4 -> 4 packed f16 and store 8B to LDS.
__device__ __forceinline__ void cvt_store4(_Float16* dst, v4f v) {
    v4h h = { (_Float16)v.x, (_Float16)v.y, (_Float16)v.z, (_Float16)v.w };
    *(v4h*)dst = h;
}

#if HAVE_ASYNC_LDS
// 16-byte async global -> LDS copy (GLOBAL_LOAD_ASYNC_TO_LDS_B128).
__device__ __forceinline__ void async_copy16(const void* gsrc, void* ldst) {
    __builtin_amdgcn_global_load_async_to_lds_b128(
        (__attribute__((address_space(1))) v4i*)gsrc,
        (__attribute__((address_space(3))) v4i*)ldst,
        0, 0);
}
#endif

// ---------------------------------------------------------------------------
// Kernel 0: zero output and per-expert counts.
// ---------------------------------------------------------------------------
__global__ void k_zero(float* __restrict__ out, int n, int* __restrict__ counts) {
    int i = blockIdx.x * 256 + threadIdx.x;
    if (i < NEXP) counts[i] = 0;
    if (i < n) out[i] = 0.0f;
}

// ---------------------------------------------------------------------------
// Kernel 1: routing. softmax -> top2 -> renormalize. Builds per-expert lists.
// ---------------------------------------------------------------------------
__global__ void k_route(const float* __restrict__ logits,
                        int* __restrict__ counts, Entry* __restrict__ lists) {
    int t = blockIdx.x * 256 + threadIdx.x;
    if (t >= TOK) return;
    float l[NEXP];
#pragma unroll
    for (int e = 0; e < NEXP; ++e) l[e] = logits[t * NEXP + e];
    int e1 = 0;
#pragma unroll
    for (int e = 1; e < NEXP; ++e) if (l[e] > l[e1]) e1 = e;   // ties -> low idx
    int e2 = (e1 == 0) ? 1 : 0;
#pragma unroll
    for (int e = 0; e < NEXP; ++e)
        if (e != e1 && l[e] > l[e2]) e2 = e;
    // renormalized top-2 softmax weights (common softmax factors cancel)
    float p2  = __expf(l[e2] - l[e1]);
    float inv = 1.0f / (1.0f + p2);
    int pos1 = atomicAdd(&counts[e1], 1);
    lists[e1 * TOK + pos1] = { t * 2 + 0, inv };
    int pos2 = atomicAdd(&counts[e2], 1);
    lists[e2 * TOK + pos2] = { t * 2 + 1, p2 * inv };
}

// ---------------------------------------------------------------------------
// Kernel 2: up-projection + SwiGLU.  act[2t+s, f] = silu(x.W1^T) * (x.W3^T)
// Block = (fchunk of 128, token-group of 32, expert); 256 threads = 8 waves.
// Wave w owns (mi = w>>2) x (n-tiles 2*(w&3), 2*(w&3)+1), gate AND up.
// ---------------------------------------------------------------------------
__global__ __launch_bounds__(256) void k_ffn1(
    const float* __restrict__ hs, const float* __restrict__ w1,
    const float* __restrict__ w3, const int* __restrict__ counts,
    const Entry* __restrict__ lists, _Float16* __restrict__ act) {
    const int fchunk = blockIdx.x;        // 0..31
    const int mg     = blockIdx.y;        // 0..63
    const int e      = blockIdx.z;        // 0..7
    const int count  = counts[e];
    if (mg * MT >= count) return;

    __shared__ _Float16 Xs[MT * XSTR];     //  8.7 KB
    __shared__ _Float16 W1s[NT * XSTR];    // 34.8 KB
    __shared__ _Float16 W3s[NT * XSTR];    // 34.8 KB
    __shared__ _Float16 Ax[MT * XSTR];     //  8.7 KB
    __shared__ int   rowAct[MT];
    __shared__ int   rowTok[MT];
    __shared__ int   rowOk[MT];

    const int t = threadIdx.x;
    if (t < MT) {
        int idx = mg * MT + t;
        if (idx < count) {
            Entry en = lists[e * TOK + idx];
            rowAct[t] = en.ts; rowTok[t] = en.ts >> 1; rowOk[t] = 1;
        } else {
            rowAct[t] = 0; rowTok[t] = 0; rowOk[t] = 0;
        }
    }
    __syncthreads();

    const int lane = t & 31;
    const int w    = t >> 5;
    const int mi   = w >> 2;          // 0..1   (M tile)
    const int nt0  = (w & 3) * 2;     // first of 2 N tiles
    const int h    = lane >> 4;       // lane half
    const int mn   = lane & 15;       // M row for A, N col for B/C

    // Hoist per-thread staging row -> token lookups out of the K loop.
    int xtok[4];
#pragma unroll
    for (int i = 0; i < 4; ++i) xtok[i] = rowTok[(t + i * 256) >> 5];

    v8f accG[2] = {};
    v8f accU[2] = {};

    const float* w1e = w1 + (size_t)e * FFN * HID + (size_t)(fchunk * NT) * HID;
    const float* w3e = w3 + (size_t)e * FFN * HID + (size_t)(fchunk * NT) * HID;

    for (int ks = 0; ks < HID; ks += KS) {
        __syncthreads();
        if (ks + KS < HID) {
            __builtin_prefetch(w1e + (size_t)(t >> 5) * HID + ks + KS + (t & 31) * 4, 0, 0);
            __builtin_prefetch(w3e + (size_t)(t >> 5) * HID + ks + KS + (t & 31) * 4, 0, 0);
        }
        // ---- stage X section: 32 rows x 128 cols (f32 -> f16), batched ----
        {
            v4f tx[4];
#pragma unroll
            for (int i = 0; i < 4; ++i) {
                int f4 = t + i * 256;                 // 0..1023
                int c4 = f4 & 31;
                tx[i] = *(const v4f*)(hs + (size_t)xtok[i] * HID + ks + c4 * 4);
            }
#pragma unroll
            for (int i = 0; i < 4; ++i) {
                int f4 = t + i * 256;
                cvt_store4(&Xs[(f4 >> 5) * XSTR + (f4 & 31) * 4], tx[i]);
            }
        }
        // ---- stage W1/W3 sections: 128 rows x 128 cols each, batched ----
#pragma unroll
        for (int i0 = 0; i0 < 16; i0 += 4) {
            v4f t1[4], t3[4];
#pragma unroll
            for (int j = 0; j < 4; ++j) {
                int f4 = t + (i0 + j) * 256;          // 0..4095
                size_t off = (size_t)(f4 >> 5) * HID + ks + (f4 & 31) * 4;
                t1[j] = *(const v4f*)(w1e + off);
                t3[j] = *(const v4f*)(w3e + off);
            }
#pragma unroll
            for (int j = 0; j < 4; ++j) {
                int f4 = t + (i0 + j) * 256;
                cvt_store4(&W1s[(f4 >> 5) * XSTR + (f4 & 31) * 4], t1[j]);
                cvt_store4(&W3s[(f4 >> 5) * XSTR + (f4 & 31) * 4], t3[j]);
            }
        }
        __syncthreads();
        // ---- 4 K-steps of 32; A reused across 4 WMMAs ----
#pragma unroll
        for (int kk = 0; kk < KS; kk += 32) {
            const _Float16* xr = Xs + (mi * 16 + mn) * XSTR + kk;
            v16h A = mkv16(*(const v8h*)(xr + 16 * h),
                           *(const v8h*)(xr + 16 * h + 8));
#pragma unroll
            for (int j = 0; j < 2; ++j) {
                const _Float16* b1 = W1s + ((nt0 + j) * 16 + mn) * XSTR + kk + 16 * h;
                v16h B1 = mkv16(*(const v8h*)(b1), *(const v8h*)(b1 + 8));
                accG[j] = wmma_f16(A, B1, accG[j]);
                const _Float16* b3 = W3s + ((nt0 + j) * 16 + mn) * XSTR + kk + 16 * h;
                v16h B3 = mkv16(*(const v8h*)(b3), *(const v8h*)(b3 + 8));
                accU[j] = wmma_f16(A, B3, accU[j]);
            }
        }
    }

    // ---- SwiGLU in registers, stage act tile to LDS ----
#pragma unroll
    for (int j = 0; j < 2; ++j) {
#pragma unroll
        for (int r = 0; r < 8; ++r) {
            int m = mi * 16 + r + 8 * h;      // C/D layout: m = r + 8*half
            float g = accG[j][r];
            float u = accU[j][r];
            float a = (g / (1.0f + __expf(-g))) * u;
            Ax[m * XSTR + (nt0 + j) * 16 + mn] = (_Float16)a;
        }
    }
    __syncthreads();
    // ---- coalesced store: 32 rows x 128 f16 (256B/row) ----
    {
        int row = t >> 3, seg = t & 7;        // 8 threads/row, 32B each
        if (rowOk[row]) {
            const _Float16* s = Ax + row * XSTR + seg * 16;
            _Float16* d = act + (size_t)rowAct[row] * FFN + fchunk * NT + seg * 16;
            *(v8h*)(d)     = *(const v8h*)(s);
            *(v8h*)(d + 8) = *(const v8h*)(s + 8);
        }
    }
}

// ---------------------------------------------------------------------------
// Kernel 3: down projection + weighted combine.
// out[tok, d] += w * act[2t+s, :] . w2[e, d, :]
// ---------------------------------------------------------------------------
__global__ __launch_bounds__(256) void k_ffn2(
    const _Float16* __restrict__ act, const float* __restrict__ w2,
    const int* __restrict__ counts, const Entry* __restrict__ lists,
    float* __restrict__ out) {
    const int dchunk = blockIdx.x;        // 0..7
    const int mg     = blockIdx.y;        // 0..63
    const int e      = blockIdx.z;
    const int count  = counts[e];
    if (mg * MT >= count) return;

    __shared__ _Float16 As[MT * XSTR];     //  8.7 KB
    __shared__ _Float16 W2s[NT * XSTR];    // 34.8 KB
    __shared__ int   rowAct[MT];
    __shared__ int   rowTok[MT];
    __shared__ float rowW[MT];
    __shared__ int   rowOk[MT];

    const int t = threadIdx.x;
    if (t < MT) {
        int idx = mg * MT + t;
        if (idx < count) {
            Entry en = lists[e * TOK + idx];
            rowAct[t] = en.ts; rowTok[t] = en.ts >> 1; rowW[t] = en.w; rowOk[t] = 1;
        } else {
            rowAct[t] = 0; rowTok[t] = 0; rowW[t] = 0.0f; rowOk[t] = 0;
        }
    }
    __syncthreads();

    const int lane = t & 31;
    const int w    = t >> 5;
    const int mi   = w >> 2;
    const int nt0  = (w & 3) * 2;
    const int h    = lane >> 4;
    const int mn   = lane & 15;

    // Hoist per-thread staging act-row lookups out of the K loop.
    int arow[2];
#pragma unroll
    for (int i = 0; i < 2; ++i) arow[i] = rowAct[(t + i * 256) >> 4];

    v8f acc[2] = {};
    const float* w2e = w2 + (size_t)e * HID * FFN + (size_t)(dchunk * NT) * FFN;

    for (int ks = 0; ks < FFN; ks += KS) {
        __syncthreads();
        // ---- kick off async act staging (overlaps with w2 conversion) ----
#if HAVE_ASYNC_LDS
#pragma unroll
        for (int i = 0; i < 2; ++i) {
            int u8 = t + i * 256;             // 0..511 (16B units)
            int c8 = u8 & 15;
            async_copy16(act + (size_t)arow[i] * FFN + ks + c8 * 8,
                         &As[(u8 >> 4) * XSTR + c8 * 8]);
        }
#else
#pragma unroll
        for (int i = 0; i < 2; ++i) {
            int u8 = t + i * 256;
            int c8 = u8 & 15;
            v8h v = *(const v8h*)(act + (size_t)arow[i] * FFN + ks + c8 * 8);
            *(v8h*)(&As[(u8 >> 4) * XSTR + c8 * 8]) = v;
        }
#endif
        if (ks + KS < FFN)
            __builtin_prefetch(w2e + (size_t)(t >> 5) * FFN + ks + KS + (t & 31) * 4, 0, 0);
        // ---- stage w2 section: 128 rows x 128 cols (f32 -> f16), batched ----
#pragma unroll
        for (int i0 = 0; i0 < 16; i0 += 4) {
            v4f t2[4];
#pragma unroll
            for (int j = 0; j < 4; ++j) {
                int f4 = t + (i0 + j) * 256;
                t2[j] = *(const v4f*)(w2e + (size_t)(f4 >> 5) * FFN + ks + (f4 & 31) * 4);
            }
#pragma unroll
            for (int j = 0; j < 4; ++j) {
                int f4 = t + (i0 + j) * 256;
                cvt_store4(&W2s[(f4 >> 5) * XSTR + (f4 & 31) * 4], t2[j]);
            }
        }
#if HAVE_ASYNC_LDS
        __builtin_amdgcn_s_wait_asynccnt(0);   // act DMA done (own wave)
#endif
        __syncthreads();
#pragma unroll
        for (int kk = 0; kk < KS; kk += 32) {
            const _Float16* ar = As + (mi * 16 + mn) * XSTR + kk;
            v16h A = mkv16(*(const v8h*)(ar + 16 * h),
                           *(const v8h*)(ar + 16 * h + 8));
#pragma unroll
            for (int j = 0; j < 2; ++j) {
                const _Float16* br = W2s + ((nt0 + j) * 16 + mn) * XSTR + kk + 16 * h;
                v16h B = mkv16(*(const v8h*)(br), *(const v8h*)(br + 8));
                acc[j] = wmma_f16(A, B, acc[j]);
            }
        }
    }

    // ---- weighted combine: exactly 2 atomics per output element ----
#pragma unroll
    for (int j = 0; j < 2; ++j) {
        const int d = dchunk * NT + (nt0 + j) * 16 + mn;
#pragma unroll
        for (int r = 0; r < 8; ++r) {
            int m = mi * 16 + r + 8 * h;
            if (rowOk[m]) {
                __hip_atomic_fetch_add(&out[(size_t)rowTok[m] * HID + d],
                                       rowW[m] * acc[j][r],
                                       __ATOMIC_RELAXED, __HIP_MEMORY_SCOPE_AGENT);
            }
        }
    }
}

// ---------------------------------------------------------------------------
extern "C" void kernel_launch(void* const* d_in, const int* in_sizes, int n_in,
                              void* d_out, int out_size, void* d_ws, size_t ws_size,
                              hipStream_t stream) {
    const float* hs     = (const float*)d_in[0];   // [2048,1024]
    const float* logits = (const float*)d_in[1];   // [2048,8]
    const float* w1     = (const float*)d_in[2];   // [8,4096,1024]
    const float* w3     = (const float*)d_in[3];   // [8,4096,1024]
    const float* w2     = (const float*)d_in[4];   // [8,1024,4096]
    float* out = (float*)d_out;                    // [2048,1024]

    char* ws = (char*)d_ws;
    int*   counts = (int*)ws;                         // 32 B
    Entry* lists  = (Entry*)(ws + 256);               // 8*2048*8 = 128 KB
    _Float16* act = (_Float16*)(ws + 262144);         // [4096][4096] f16 = 32 MB

    k_zero <<<dim3((TOK * HID + 255) / 256), 256, 0, stream>>>(out, TOK * HID, counts);
    k_route<<<dim3(TOK / 256),               256, 0, stream>>>(logits, counts, lists);
    k_ffn1 <<<dim3(FFN / NT, TOK / MT, NEXP), 256, 0, stream>>>(hs, w1, w3, counts, lists, act);
    k_ffn2 <<<dim3(HID / NT, TOK / MT, NEXP), 256, 0, stream>>>(act, w2, counts, lists, out);
}
